// Block_20796231647272
// MI455X (gfx1250) — matrix operand order repaired and checked by reading, not compiled, not run on hardware
//
#include <hip/hip_runtime.h>
#include <math.h>

typedef __bf16 bf16;
typedef __attribute__((ext_vector_type(16))) __bf16 v16bf;
typedef __attribute__((ext_vector_type(8)))  __bf16 bf16x8;
typedef __attribute__((ext_vector_type(8)))  float  v8f;
typedef __attribute__((ext_vector_type(4)))  int    v4i;

#define LDST 40  // padded LDS row stride (bf16 elems) for a 32-wide K tile

// ---- CDNA5 async global->LDS path (guarded; falls back to reg round-trip) ----
#ifndef HAVE_ASYNC_LDS
#if defined(__has_builtin)
#if __has_builtin(__builtin_amdgcn_global_load_async_to_lds_b128)
#define HAVE_ASYNC_LDS 1
#endif
#endif
#endif
#ifndef HAVE_ASYNC_LDS
#define HAVE_ASYNC_LDS 0
#endif

#define AS1 __attribute__((address_space(1)))
#define AS3 __attribute__((address_space(3)))

#if HAVE_ASYNC_LDS
__device__ __forceinline__ void async_ld128(const bf16* gsrc, bf16* ldst) {
  __builtin_amdgcn_global_load_async_to_lds_b128((AS1 v4i*)gsrc, (AS3 v4i*)ldst,
                                                 /*offset=*/0, /*cpol=*/0);
}
__device__ __forceinline__ void wait_async0() {
#if defined(__has_builtin)
#if __has_builtin(__builtin_amdgcn_s_wait_asynccnt)
  __builtin_amdgcn_s_wait_asynccnt(0);
#else
  asm volatile("s_wait_asynccnt 0x0" ::: "memory");
#endif
#else
  asm volatile("s_wait_asynccnt 0x0" ::: "memory");
#endif
}
#endif

// ---------------------------------------------------------------------------
// f32 -> bf16 conversion (weights / activations)
// ---------------------------------------------------------------------------
__launch_bounds__(256)
__global__ void f2bf_kernel(const float* __restrict__ in, bf16* __restrict__ out, size_t n) {
  size_t i = (size_t)blockIdx.x * 256 + threadIdx.x;
  if (i < n) out[i] = (bf16)in[i];
}

// ---------------------------------------------------------------------------
// Fused LayerNorm + time-shift + token-mix.  One wave32 per (b,t) row, C=512.
// ---------------------------------------------------------------------------
__launch_bounds__(256)
__global__ void ln_mix_kernel(const float* __restrict__ x,
                              const float* __restrict__ gamma, const float* __restrict__ beta,
                              const float* __restrict__ mk, const float* __restrict__ mv,
                              const float* __restrict__ mr,
                              bf16* __restrict__ ok, bf16* __restrict__ ov, bf16* __restrict__ orr,
                              int T, int haveV)
{
  const int C = 512;
  const int wave = threadIdx.x >> 5;
  const int lane = threadIdx.x & 31;
  const int row  = blockIdx.x * 8 + wave;            // b*T + t
  const int t    = row & (T - 1);
  const float* xr = x + (size_t)row * C;
  const float* xp = xr - C;

  float cur[16], prv[16];
  float s0 = 0.f, q0 = 0.f, s1 = 0.f, q1 = 0.f;
#pragma unroll
  for (int j = 0; j < 16; ++j) {
    int c = lane + 32 * j;
    float a = xr[c];
    cur[j] = a; s0 += a; q0 += a * a;
    float b = (t > 0) ? xp[c] : 0.f;
    prv[j] = b; s1 += b; q1 += b * b;
  }
#pragma unroll
  for (int off = 16; off > 0; off >>= 1) {
    s0 += __shfl_xor(s0, off);
    q0 += __shfl_xor(q0, off);
    s1 += __shfl_xor(s1, off);
    q1 += __shfl_xor(q1, off);
  }
  const float inv = 1.f / (float)C;
  float m0 = s0 * inv, v0 = q0 * inv - m0 * m0;
  float r0 = rsqrtf(v0 + 1e-5f);
  float m1 = s1 * inv, v1 = q1 * inv - m1 * m1;
  float r1 = rsqrtf(v1 + 1e-5f);
#pragma unroll
  for (int j = 0; j < 16; ++j) {
    int c = lane + 32 * j;
    float g = gamma[c], be = beta[c];
    float xn = (cur[j] - m0) * r0 * g + be;
    float xs = (t > 0) ? ((prv[j] - m1) * r1 * g + be) : 0.f;
    size_t idx = (size_t)row * C + c;
    float ck = mk[c];
    ok[idx] = (bf16)(xn * ck + xs * (1.f - ck));
    if (haveV) {
      float cv = mv[c];
      ov[idx] = (bf16)(xn * cv + xs * (1.f - cv));
    }
    float cr = mr[c];
    orr[idx] = (bf16)(xn * cr + xs * (1.f - cr));
  }
}

// ---------------------------------------------------------------------------
// WMMA GEMM:  O[M,N] (f32) = A[M,K] (bf16, row-major) * W[N,K]^T (bf16, row-major)
// Block tile 128x128, K-step 32.  8 waves; each wave: 2x4 WMMA 16x16x32 tiles.
// N, K compile-time -> immediate store offsets.  Async-to-LDS double-buffered
// pipeline (tile k+1 DMA overlapped with compute, one barrier per K-step);
// falls back to reg round-trip with two barriers if builtin is missing.
// ---------------------------------------------------------------------------
template <int N, int K>
__launch_bounds__(256)
__global__ void gemm_bf16_kernel(const bf16* __restrict__ A,
                                 const bf16* __restrict__ W,
                                 float* __restrict__ O,
                                 int M)
{
  constexpr int KT = K / 32;
  __shared__ __align__(16) bf16 sA[2][128 * LDST];
  __shared__ __align__(16) bf16 sB[2][128 * LDST];

  const int tid  = threadIdx.x;
  const int wave = tid >> 5;
  const int lane = tid & 31;
  const int g    = lane >> 4;      // half-wave group (K interleave select)
  const int lr   = lane & 15;      // row/col within 16-wide fragment
  const int m0   = blockIdx.y * 128;
  const int n0   = blockIdx.x * 128;
  const int wm   = (wave >> 1) * 32;   // wave M offset inside block tile
  const int wn   = (wave & 1) * 64;    // wave N offset inside block tile

  v8f acc[2][4];
#pragma unroll
  for (int i = 0; i < 2; ++i)
#pragma unroll
    for (int j = 0; j < 4; ++j)
#pragma unroll
      for (int e = 0; e < 8; ++e) acc[i][j][e] = 0.f;

  // cooperative tile-load mapping: thread -> (row 0..127, 16-elem half)
  const int lm = tid >> 1;
  const int lk = (tid & 1) * 16;
  const bf16* Aload = A + (size_t)(m0 + lm) * K + lk;
  const bf16* Wload = W + (size_t)(n0 + lm) * K + lk;
  const int sOff = lm * LDST + lk;

#if HAVE_ASYNC_LDS
  // prologue: DMA tile 0 into buffer 0
  async_ld128(Aload,     &sA[0][sOff]);
  async_ld128(Aload + 8, &sA[0][sOff + 8]);
  async_ld128(Wload,     &sB[0][sOff]);
  async_ld128(Wload + 8, &sB[0][sOff + 8]);
#endif

  for (int kb = 0; kb < KT; ++kb) {
    const int cur = kb & 1;
#if HAVE_ASYNC_LDS
    wait_async0();                       // own DMA for tile kb done
    __syncthreads();                     // everyone's DMA done, prior reads of other buffer done
    if (kb + 1 < KT) {                   // DMA tile kb+1 into other buffer (overlaps compute)
      const bf16* ap = Aload + ((kb + 1) << 5);
      const bf16* wp = Wload + ((kb + 1) << 5);
      async_ld128(ap,     &sA[cur ^ 1][sOff]);
      async_ld128(ap + 8, &sA[cur ^ 1][sOff + 8]);
      async_ld128(wp,     &sB[cur ^ 1][sOff]);
      async_ld128(wp + 8, &sB[cur ^ 1][sOff + 8]);
    }
#else
    const bf16* ap = Aload + (kb << 5);
    const bf16* wp = Wload + (kb << 5);
    bf16x8 a0 = *(const bf16x8*)(ap);
    bf16x8 a1 = *(const bf16x8*)(ap + 8);
    bf16x8 b0 = *(const bf16x8*)(wp);
    bf16x8 b1 = *(const bf16x8*)(wp + 8);
    if (kb + 1 < KT) {
      __builtin_prefetch(ap + 32, 0, 1);
      __builtin_prefetch(wp + 32, 0, 1);
    }
    __syncthreads();
    *(bf16x8*)(&sA[cur][sOff])     = a0;
    *(bf16x8*)(&sA[cur][sOff + 8]) = a1;
    *(bf16x8*)(&sB[cur][sOff])     = b0;
    *(bf16x8*)(&sB[cur][sOff + 8]) = b1;
    __syncthreads();
#endif

    // A fragments: 16x32 bf16 layout — lane<16 holds K{8g..8g+7, 16+8g..16+8g+7}
    v16bf af[2];
#pragma unroll
    for (int i = 0; i < 2; ++i) {
      const bf16* rowp = &sA[cur][(wm + i * 16 + lr) * LDST];
      *(bf16x8*)&af[i]       = *(const bf16x8*)(rowp + 8 * g);
      *((bf16x8*)&af[i] + 1) = *(const bf16x8*)(rowp + 16 + 8 * g);
    }
    // B fragments: 32x16 bf16 layout — lane half g holds K{8g..8g+7, 16+8g..16+8g+7}
    v16bf bfr[4];
#pragma unroll
    for (int j = 0; j < 4; ++j) {
      const bf16* rowp = &sB[cur][(wn + j * 16 + lr) * LDST];
      *(bf16x8*)&bfr[j]       = *(const bf16x8*)(rowp + 8 * g);
      *((bf16x8*)&bfr[j] + 1) = *(const bf16x8*)(rowp + 16 + 8 * g);
    }
#pragma unroll
    for (int i = 0; i < 2; ++i)
#pragma unroll
      for (int j = 0; j < 4; ++j)
        acc[i][j] = __builtin_amdgcn_wmma_f32_16x16x32_bf16(
            false, af[i], false, bfr[j], (short)0, acc[i][j], false, false);
  }

  // C/D layout: VGPR r -> M = 8*g + r, N = lr.  N compile-time => immediate offsets.
  float* op0 = O + (size_t)(m0 + wm + 8 * g) * N + (n0 + wn + lr);
#pragma unroll
  for (int i = 0; i < 2; ++i)
#pragma unroll
    for (int j = 0; j < 4; ++j)
#pragma unroll
      for (int r = 0; r < 8; ++r)
        op0[(size_t)(i * 16 + r) * N + j * 16] = acc[i][j][r];
}

// ---------------------------------------------------------------------------
// Numerically-stable WKV recurrence.  One thread per (b,c) channel.
// ---------------------------------------------------------------------------
__launch_bounds__(256)
__global__ void wkv_kernel(const float* __restrict__ td, const float* __restrict__ tf,
                           const float* __restrict__ k, const float* __restrict__ v,
                           float* __restrict__ y, int T)
{
  const int C = 512;
  int idx = blockIdx.x * 256 + threadIdx.x;   // b*C + c
  int c = idx & (C - 1);
  int b = idx >> 9;
  const float w = -expf(td[c]);
  const float u = tf[c];
  size_t off = (size_t)b * T * C + c;
  float p = 0.f, q = 0.f, o = -1e38f;
  for (int t = 0; t < T; ++t, off += C) {
    float kt = k[off], vt = v[off];
    float no = fmaxf(o, u + kt);
    float Ae = expf(o - no);
    float Bx = expf(u + kt - no);
    y[off] = (Ae * p + Bx * vt) / (Ae * q + Bx);
    float no2 = fmaxf(o + w, kt);
    float A2 = expf(o + w - no2);
    float B2 = expf(kt - no2);
    p = A2 * p + B2 * vt;
    q = A2 * q + B2;
    o = no2;
  }
}

// ---------------------------------------------------------------------------
// Elementwise epilogues
// ---------------------------------------------------------------------------
__launch_bounds__(256)
__global__ void sigmul_bf16_kernel(const float* __restrict__ r, const float* __restrict__ wkv,
                                   bf16* __restrict__ out, size_t n) {
  size_t i = (size_t)blockIdx.x * 256 + threadIdx.x;
  if (i < n) {
    float s = 1.f / (1.f + expf(-r[i]));
    out[i] = (bf16)(s * wkv[i]);
  }
}

__launch_bounds__(256)
__global__ void add_kernel(const float* __restrict__ a, const float* __restrict__ b,
                           float* __restrict__ o, size_t n) {
  size_t i = (size_t)blockIdx.x * 256 + threadIdx.x;
  if (i < n) o[i] = a[i] + b[i];
}

__launch_bounds__(256)
__global__ void relu2_bf16_kernel(const float* __restrict__ in, bf16* __restrict__ out, size_t n) {
  size_t i = (size_t)blockIdx.x * 256 + threadIdx.x;
  if (i < n) {
    float v = fmaxf(in[i], 0.f);
    out[i] = (bf16)(v * v);
  }
}

__launch_bounds__(256)
__global__ void final_kernel(const float* __restrict__ x1, const float* __restrict__ rr,
                             const float* __restrict__ kv, float* __restrict__ o, size_t n) {
  size_t i = (size_t)blockIdx.x * 256 + threadIdx.x;
  if (i < n) {
    float s = 1.f / (1.f + expf(-rr[i]));
    o[i] = x1[i] + s * kv[i];
  }
}

// ---------------------------------------------------------------------------
// Host launcher
// ---------------------------------------------------------------------------
static inline unsigned blocks_for(size_t n) { return (unsigned)((n + 255) / 256); }

extern "C" void kernel_launch(void* const* d_in, const int* in_sizes, int n_in,
                              void* d_out, int out_size, void* d_ws, size_t ws_size,
                              hipStream_t stream) {
  const int B = 32, T = 1024, C = 512, H = 2048;
  const size_t N1 = (size_t)B * T * C;   // 16.78M
  const size_t NH = (size_t)B * T * H;   // 67.1M

  const float* x     = (const float*)d_in[0];
  const float* ln1_g = (const float*)d_in[1];
  const float* ln1_b = (const float*)d_in[2];
  const float* ln2_g = (const float*)d_in[3];
  const float* ln2_b = (const float*)d_in[4];
  const float* tdec  = (const float*)d_in[5];
  const float* tfir  = (const float*)d_in[6];
  const float* tm_k  = (const float*)d_in[7];
  const float* tm_v  = (const float*)d_in[8];
  const float* tm_r  = (const float*)d_in[9];
  const float* Wk    = (const float*)d_in[10];
  const float* Wv    = (const float*)d_in[11];
  const float* Wr    = (const float*)d_in[12];
  const float* Wo    = (const float*)d_in[13];
  const float* cm_k  = (const float*)d_in[14];
  const float* cm_r  = (const float*)d_in[15];
  const float* cWk   = (const float*)d_in[16];
  const float* cWr   = (const float*)d_in[17];
  const float* cWv   = (const float*)d_in[18];

  char* p = (char*)d_ws;
  auto alloc = [&](size_t bytes) -> char* {
    char* r = p;
    p += (bytes + 255) & ~(size_t)255;
    return r;
  };

  // bf16 weights
  bf16* bWk  = (bf16*)alloc((size_t)C * C * 2);
  bf16* bWv  = (bf16*)alloc((size_t)C * C * 2);
  bf16* bWr  = (bf16*)alloc((size_t)C * C * 2);
  bf16* bWo  = (bf16*)alloc((size_t)C * C * 2);
  bf16* bcWk = (bf16*)alloc((size_t)H * C * 2);
  bf16* bcWr = (bf16*)alloc((size_t)C * C * 2);
  bf16* bcWv = (bf16*)alloc((size_t)C * H * 2);
  // bf16 activations
  bf16* bxk  = (bf16*)alloc(N1 * 2);   // later reused as sigmoid(r)*wkv
  bf16* bxv  = (bf16*)alloc(N1 * 2);   // later reused as xk2
  bf16* bxr  = (bf16*)alloc(N1 * 2);   // later reused as xr2
  bf16* bkk  = (bf16*)alloc(NH * 2);
  // f32 buffers
  float* fk   = (float*)alloc(N1 * 4); // later reused as att
  float* fv   = (float*)alloc(N1 * 4); // later reused as x1
  float* fr   = (float*)alloc(N1 * 4); // later reused as kv
  float* fwkv = (float*)alloc(N1 * 4); // later reused as rr
  float* fkk  = (float*)alloc(NH * 4);

  float* out = (float*)d_out;
  const size_t CC = (size_t)C * C;
  const size_t HC = (size_t)H * C;

  // 1) weights -> bf16
  f2bf_kernel<<<blocks_for(CC), 256, 0, stream>>>(Wk,  bWk,  CC);
  f2bf_kernel<<<blocks_for(CC), 256, 0, stream>>>(Wv,  bWv,  CC);
  f2bf_kernel<<<blocks_for(CC), 256, 0, stream>>>(Wr,  bWr,  CC);
  f2bf_kernel<<<blocks_for(CC), 256, 0, stream>>>(Wo,  bWo,  CC);
  f2bf_kernel<<<blocks_for(HC), 256, 0, stream>>>(cWk, bcWk, HC);
  f2bf_kernel<<<blocks_for(CC), 256, 0, stream>>>(cWr, bcWr, CC);
  f2bf_kernel<<<blocks_for(HC), 256, 0, stream>>>(cWv, bcWv, HC);

  const int rows = B * T;              // 32768
  const int M = rows;

  // 2) LN1 + shift + mix -> bf16 xk, xv, xr
  ln_mix_kernel<<<rows / 8, 256, 0, stream>>>(x, ln1_g, ln1_b, tm_k, tm_v, tm_r,
                                              bxk, bxv, bxr, T, 1);

  // 3-5) k, v, r projections  (N=512, K=512)
  gemm_bf16_kernel<512, 512><<<dim3(C / 128, M / 128), 256, 0, stream>>>(bxk, bWk, fk, M);
  gemm_bf16_kernel<512, 512><<<dim3(C / 128, M / 128), 256, 0, stream>>>(bxv, bWv, fv, M);
  gemm_bf16_kernel<512, 512><<<dim3(C / 128, M / 128), 256, 0, stream>>>(bxr, bWr, fr, M);

  // 6) WKV scan
  wkv_kernel<<<(B * C) / 256, 256, 0, stream>>>(tdec, tfir, fk, fv, fwkv, T);

  // 7) sigmoid(r) * wkv -> bf16 (reuse bxk)
  bf16* bsr = bxk;
  sigmul_bf16_kernel<<<blocks_for(N1), 256, 0, stream>>>(fr, fwkv, bsr, N1);

  // 8) output projection (reuse fk as att)
  float* fatt = fk;
  gemm_bf16_kernel<512, 512><<<dim3(C / 128, M / 128), 256, 0, stream>>>(bsr, bWo, fatt, M);

  // 9) residual (reuse fv as x1)
  float* fx1 = fv;
  add_kernel<<<blocks_for(N1), 256, 0, stream>>>(x, fatt, fx1, N1);

  // 10) LN2 + shift + mix -> bf16 xk2, xr2 (reuse bxv, bxr)
  bf16* bxk2 = bxv;
  bf16* bxr2 = bxr;
  ln_mix_kernel<<<rows / 8, 256, 0, stream>>>(fx1, ln2_g, ln2_b, cm_k, nullptr, cm_r,
                                              bxk2, nullptr, bxr2, T, 0);

  // 11) kk = xk2 @ cWk^T   [M, H]  (N=2048, K=512)
  gemm_bf16_kernel<2048, 512><<<dim3(H / 128, M / 128), 256, 0, stream>>>(bxk2, bcWk, fkk, M);

  // 12) relu(kk)^2 -> bf16
  relu2_bf16_kernel<<<blocks_for(NH), 256, 0, stream>>>(fkk, bkk, NH);

  // 13) kv = kk2 @ cWv^T   [M, C]  (N=512, K=2048; reuse fr as kv)
  float* fkv = fr;
  gemm_bf16_kernel<512, 2048><<<dim3(C / 128, M / 128), 256, 0, stream>>>(bkk, bcWv, fkv, M);

  // 14) rr = xr2 @ cWr^T   (reuse fwkv as rr)
  float* frr = fwkv;
  gemm_bf16_kernel<512, 512><<<dim3(C / 128, M / 128), 256, 0, stream>>>(bxr2, bcWr, frr, M);

  // 15) out = x1 + sigmoid(rr) * kv
  final_kernel<<<blocks_for(N1), 256, 0, stream>>>(fx1, frr, fkv, out, N1);
}
